// SemanticGuidedOT_50371376448007
// MI455X (gfx1250) — matrix operand order repaired and checked by reading, not compiled, not run on hardware
//
#include <hip/hip_runtime.h>
#include <hip/hip_bf16.h>

// ---------------------------------------------------------------------------
// SemanticGuidedOT for MI455X (gfx1250, wave32, WMMA f16)
//
// Key exact simplifications vs the reference:
//  * kv-length-1 attention: softmax == 1, attn_out is constant over N,
//    wq/wk are dead.  attn_const[b,c] = wo @ (wv @ txt_emb + bv) + bo.
//  * cos(img_n, txt_n) == sn . txt_n  (same per-pixel l2 norm).
//  * sim (= sn sn^T) is symmetric -> K symmetric -> both Sinkhorn matvecs are
//    row reductions, and T^T @ source reads K row-major (contiguous f16).
//  * K stored f16 (values in [e^-40,1]): 67 MB total -> resident in 192MB L2.
// ---------------------------------------------------------------------------

typedef __attribute__((ext_vector_type(16))) _Float16 v16h;
typedef __attribute__((ext_vector_type(8)))  float    v8f;
typedef _Float16 f16;

#define NPIX 4096      // H*W
#define CCH  256       // channels
#define BB   2         // batch
#define K2SCALE 28.853900817779268f   // log2(e)/EPS_OT, EPS_OT=0.05

union AV { v16h v; unsigned int u[8]; };

// ---------------- block reduction helpers (blockDim.x == 256) --------------
__device__ __forceinline__ float block_sum256(float v, float* red) {
  int t = threadIdx.x;
  red[t] = v; __syncthreads();
  #pragma unroll
  for (int s = 128; s > 0; s >>= 1) {
    if (t < s) red[t] += red[t + s];
    __syncthreads();
  }
  float r = red[0]; __syncthreads();
  return r;
}
__device__ __forceinline__ float block_max256(float v, float* red) {
  int t = threadIdx.x;
  red[t] = v; __syncthreads();
  #pragma unroll
  for (int s = 128; s > 0; s >>= 1) {
    if (t < s) red[t] = fmaxf(red[t], red[t + s]);
    __syncthreads();
  }
  float r = red[0]; __syncthreads();
  return r;
}

// ---------------------------------------------------------------------------
// 1) text pipeline: txt_emb -> txt_n, and constant attention output
//    attn_const = wo @ (wv @ txt_emb + bv) + bo          grid = B, block = 256
// ---------------------------------------------------------------------------
__global__ __launch_bounds__(256)
void text_kernel(const float* __restrict__ txt_feat, const float* __restrict__ tp_w,
                 const float* __restrict__ tp_b,
                 const float* __restrict__ wv, const float* __restrict__ bv,
                 const float* __restrict__ wo, const float* __restrict__ bo,
                 float* __restrict__ txt_n, float* __restrict__ attn_const) {
  int b = blockIdx.x, c = threadIdx.x;
  __shared__ float emb[CCH];
  __shared__ float red[CCH];
  const float* tf = txt_feat + b * 512;
  const float* wr = tp_w + (size_t)c * 512;
  float acc = tp_b[c];
  for (int j = 0; j < 512; ++j) acc += tf[j] * wr[j];
  emb[c] = acc;
  __syncthreads();
  float ss = block_sum256(acc * acc, red);
  float inv = 1.0f / fmaxf(sqrtf(ss), 1e-12f);
  txt_n[b * CCH + c] = acc * inv;
  // v = wv @ emb + bv
  float vacc = bv[c];
  const float* wvr = wv + (size_t)c * CCH;
  for (int j = 0; j < CCH; ++j) vacc += wvr[j] * emb[j];
  __syncthreads();
  red[c] = vacc;
  __syncthreads();
  // attn_const = wo @ v + bo
  float oacc = bo[c];
  const float* wor = wo + (size_t)c * CCH;
  for (int j = 0; j < CCH; ++j) oacc += wor[j] * red[j];
  attn_const[b * CCH + c] = oacc;
}

// ---------------------------------------------------------------------------
// 2) per-pixel: x = img + attn_const; LayerNorm -> source; row l2-norm -> sn;
//    cos = sn . txt_n.   Writes sn f16 [b,n,c], sourceT f16 [b,c,n], cos.
//    grid = B*N, block = 256 (one thread per channel)
// ---------------------------------------------------------------------------
__global__ __launch_bounds__(256)
void source_kernel(const float* __restrict__ img, const float* __restrict__ attn_const,
                   const float* __restrict__ ln_g, const float* __restrict__ ln_b,
                   const float* __restrict__ txt_n,
                   f16* __restrict__ sn, f16* __restrict__ srcT,
                   float* __restrict__ cosb) {
  int n = blockIdx.x & (NPIX - 1);
  int b = blockIdx.x >> 12;
  int c = threadIdx.x;
  __shared__ float red[CCH];
  float x = img[((size_t)b * CCH + c) * NPIX + n] + attn_const[b * CCH + c];
  float mu  = block_sum256(x, red) * (1.0f / CCH);
  float ex2 = block_sum256(x * x, red) * (1.0f / CCH);
  float var = ex2 - mu * mu;
  float y = (x - mu) * rsqrtf(var + 1e-5f) * ln_g[c] + ln_b[c];
  float ss = block_sum256(y * y, red);
  float inv = 1.0f / fmaxf(sqrtf(ss), 1e-12f);
  float s = y * inv;
  sn[((size_t)b * NPIX + n) * CCH + c] = (f16)s;
  srcT[((size_t)b * CCH + c) * NPIX + n] = (f16)y;
  float cv = block_sum256(s * txt_n[b * CCH + c], red);
  if (c == 0) cosb[b * NPIX + n] = cv;
}

// ---------------------------------------------------------------------------
// 3) marginals: q = norm(relu(density)+1e-6), p = softmax(cos / temp)
//    grid = B, block = 256
// ---------------------------------------------------------------------------
__global__ __launch_bounds__(256)
void marginals_kernel(const float* __restrict__ density, const float* __restrict__ cosb,
                      const float* __restrict__ temperature,
                      float* __restrict__ q_marg, float* __restrict__ p_marg) {
  int b = blockIdx.x, t = threadIdx.x;
  __shared__ float red[CCH];
  const float* db = density + (size_t)b * NPIX;
  float s = 0.f;
  for (int i = t; i < NPIX; i += 256) s += fmaxf(db[i], 0.f) + 1e-6f;
  float qs = block_sum256(s, red);
  float invq = 1.0f / qs;
  for (int i = t; i < NPIX; i += 256)
    q_marg[(size_t)b * NPIX + i] = (fmaxf(db[i], 0.f) + 1e-6f) * invq;

  float invT = 1.0f / fmaxf(temperature[0], 0.01f);
  const float* cb = cosb + (size_t)b * NPIX;
  float m = -1e30f;
  for (int i = t; i < NPIX; i += 256) m = fmaxf(m, cb[i] * invT);
  float mx = block_max256(m, red);
  float es = 0.f;
  for (int i = t; i < NPIX; i += 256) es += expf(cb[i] * invT - mx);
  float Z = block_sum256(es, red);
  float invZ = 1.0f / Z;
  for (int i = t; i < NPIX; i += 256)
    p_marg[(size_t)b * NPIX + i] = expf(cb[i] * invT - mx) * invZ;
}

// ---------------------------------------------------------------------------
// 4) WMMA GEMM #1:  K[i,j] = exp2( (sn_i . sn_j - 1) * log2(e)/eps ), f16 out.
//    Per wave: one 16-row i-tile (A held in regs, K-dim C=256 = 8 chunks),
//    swept over 16 j-tiles.  wave tasks = B * 256 * 16 ; 8 waves / block.
// ---------------------------------------------------------------------------
__global__ __launch_bounds__(256)
void sim_exp_kernel(const f16* __restrict__ sn, f16* __restrict__ Km) {
  int wave = blockIdx.x * 8 + (threadIdx.x >> 5);
  int jg = wave & 15;
  int it = (wave >> 4) & 255;
  int b  = wave >> 12;
  int lane = threadIdx.x & 31;
  int g  = lane >> 4;     // half-wave group
  int mr = lane & 15;     // A row / B column / D column
  const f16* snb = sn + (size_t)b * NPIX * CCH;
  // preload full-K A operand (8 chunks of 16x32 f16)
  AV a[8];
  const f16* arow = snb + (size_t)(it * 16 + mr) * CCH;
  #pragma unroll
  for (int kc = 0; kc < 8; ++kc) {
    #pragma unroll
    for (int v = 0; v < 8; ++v) {
      int k = kc * 32 + ((v >> 2) << 4) + g * 8 + ((v & 3) << 1);
      a[kc].u[v] = *reinterpret_cast<const unsigned int*>(arow + k);
    }
  }
  f16* kout = Km + (size_t)b * NPIX * NPIX;
  for (int jt = 0; jt < 16; ++jt) {
    int j0 = (jg * 16 + jt) * 16;
    const f16* brow = snb + (size_t)(j0 + mr) * CCH;  // column mr of B tile
    v8f c = {0.f, 0.f, 0.f, 0.f, 0.f, 0.f, 0.f, 0.f};
    #pragma unroll
    for (int kc = 0; kc < 8; ++kc) {
      AV bm;
      #pragma unroll
      for (int v = 0; v < 8; ++v) {
        int k = kc * 32 + ((v >> 2) << 4) + g * 8 + ((v & 3) << 1);
        bm.u[v] = *reinterpret_cast<const unsigned int*>(brow + k);
      }
      c = __builtin_amdgcn_wmma_f32_16x16x32_f16(false, a[kc].v, false, bm.v,
                                                 (short)0, c, false, false);
    }
    #pragma unroll
    for (int r = 0; r < 8; ++r) {
      int row = it * 16 + r + 8 * g;
      float e = exp2f((c[r] - 1.0f) * K2SCALE);
      kout[(size_t)row * NPIX + j0 + mr] = (f16)e;
    }
  }
}

// ---------------------------------------------------------------------------
// 5) Sinkhorn half-step (K symmetric):  y[i] = num[i] / (sum_j K[i,j]*x[j]+1e-8)
//    grid = (N/8, B), block = 256 (8 waves, one row per wave); x staged in LDS.
// ---------------------------------------------------------------------------
__global__ __launch_bounds__(256)
void sinkhorn_step(const f16* __restrict__ Km, const float* __restrict__ x,
                   const float* __restrict__ num, float* __restrict__ y) {
  __shared__ float xs[NPIX];
  int b = blockIdx.y;
  const float* xb = x + (size_t)b * NPIX;
  for (int i = threadIdx.x; i < NPIX; i += 256) xs[i] = xb[i];
  __syncthreads();
  int w = threadIdx.x >> 5, lane = threadIdx.x & 31;
  int row = blockIdx.x * 8 + w;
  const f16* kr = Km + (size_t)b * NPIX * NPIX + (size_t)row * NPIX;
  float acc = 0.f;
  for (int j = lane * 2; j < NPIX; j += 64) {
    union { unsigned int u; f16 h[2]; } p;
    p.u = *reinterpret_cast<const unsigned int*>(kr + j);
    acc += (float)p.h[0] * xs[j] + (float)p.h[1] * xs[j + 1];
  }
  #pragma unroll
  for (int off = 16; off > 0; off >>= 1) acc += __shfl_down(acc, off, 32);
  if (lane == 0)
    y[(size_t)b * NPIX + row] = num[(size_t)b * NPIX + row] / (acc + 1e-8f);
}

__global__ void fill_kernel(float* __restrict__ p, float v, int n) {
  int i = blockIdx.x * 256 + threadIdx.x;
  if (i < n) p[i] = v;
}

// 6) in-place column scale:  K[b,i,j] *= u[b,j]   (K symmetric: K[i,j]u[j] == K[j,i]u[j])
__global__ __launch_bounds__(256)
void scale_k_kernel(f16* __restrict__ Km, const float* __restrict__ u, size_t total) {
  size_t i = (size_t)blockIdx.x * 256 + threadIdx.x;
  size_t stride = (size_t)gridDim.x * 256;
  for (; i < total; i += stride) {
    size_t b = i >> 24;                    // N*N = 2^24
    int j = (int)(i & (NPIX - 1));
    Km[i] = (f16)((float)Km[i] * u[b * NPIX + j]);
  }
}

// ---------------------------------------------------------------------------
// 7) WMMA GEMM #2: fused[m,c] = vv[m] * sum_n Ks[m,n] * source[n,c]
//    A = Ks rows (k over n, contiguous f16 pairs); B = srcT rows (col=c, k=n).
//    wave tasks = B * (N/16) * (C/16) = 8192 ; 128 K-chunks of 32.
// ---------------------------------------------------------------------------
__global__ __launch_bounds__(256)
void fused_gemm_kernel(const f16* __restrict__ Ks, const f16* __restrict__ srcT,
                       const float* __restrict__ vv, float* __restrict__ fused) {
  int wave = blockIdx.x * 8 + (threadIdx.x >> 5);
  int ct = wave & 15;
  int mt = (wave >> 4) & 255;
  int b  = wave >> 12;
  int lane = threadIdx.x & 31;
  int g  = lane >> 4;
  int lr = lane & 15;
  const f16* arow = Ks   + (size_t)b * NPIX * NPIX + (size_t)(mt * 16 + lr) * NPIX;
  const f16* brow = srcT + (size_t)b * CCH  * NPIX + (size_t)(ct * 16 + lr) * NPIX;
  v8f c = {0.f, 0.f, 0.f, 0.f, 0.f, 0.f, 0.f, 0.f};
  for (int kc = 0; kc < 128; ++kc) {
    int kb = kc * 32;
    __builtin_prefetch(arow + kb + 256, 0, 1);
    AV a, bm;
    #pragma unroll
    for (int v = 0; v < 8; ++v) {
      int ka = kb + ((v >> 2) << 4) + g * 8 + ((v & 3) << 1);
      a.u[v] = *reinterpret_cast<const unsigned int*>(arow + ka);
      int kk = kb + ((v >> 2) << 4) + g * 8 + ((v & 3) << 1);
      bm.u[v] = *reinterpret_cast<const unsigned int*>(brow + kk);
    }
    c = __builtin_amdgcn_wmma_f32_16x16x32_f16(false, a.v, false, bm.v,
                                               (short)0, c, false, false);
  }
  #pragma unroll
  for (int r = 0; r < 8; ++r) {
    int row = mt * 16 + r + 8 * g;
    fused[((size_t)b * NPIX + row) * CCH + ct * 16 + lr] =
        c[r] * vv[(size_t)b * NPIX + row];
  }
}

// ---------------------------------------------------------------------------
// 8) out_feat = img + 1x1 conv(fused^T)   grid = B*N, block = 256
// ---------------------------------------------------------------------------
__global__ __launch_bounds__(256)
void proj_add_kernel(const float* __restrict__ img, const float* __restrict__ fused,
                     const float* __restrict__ op_w, const float* __restrict__ op_b,
                     float* __restrict__ out) {
  int n = blockIdx.x & (NPIX - 1);
  int b = blockIdx.x >> 12;
  int c = threadIdx.x;
  __shared__ float row[CCH];
  row[c] = fused[((size_t)b * NPIX + n) * CCH + c];
  __syncthreads();
  const float* wr = op_w + (size_t)c * CCH;
  float acc = op_b[c];
  for (int j = 0; j < CCH; ++j) acc += wr[j] * row[j];
  out[((size_t)b * CCH + c) * NPIX + n] = img[((size_t)b * CCH + c) * NPIX + n] + acc;
}

// ---------------------------------------------------------------------------
// 9) 3x3 conv (256 -> 128), LDS-tiled 16x16 spatial tiles with halo.
//    grid = (16 tiles, 128 outch, B), block = 256
// ---------------------------------------------------------------------------
__global__ __launch_bounds__(256)
void conv3_kernel(const float* __restrict__ in, const float* __restrict__ w,
                  const float* __restrict__ bias, float* __restrict__ out) {
  int tile = blockIdx.x, o = blockIdx.y, b = blockIdx.z;
  int ty0 = (tile >> 2) * 16, tx0 = (tile & 3) * 16;
  int tx = threadIdx.x & 15, ty = threadIdx.x >> 4;
  __shared__ float smem[18 * 18];
  float acc = bias[o];
  const float* wb = w + (size_t)o * CCH * 9;
  for (int cc = 0; cc < CCH; ++cc) {
    const float* inc = in + ((size_t)b * CCH + cc) * NPIX;
    for (int idx = threadIdx.x; idx < 324; idx += 256) {
      int ly = idx / 18, lx = idx - ly * 18;
      int gy = ty0 + ly - 1, gx = tx0 + lx - 1;
      float v = 0.f;
      if (gy >= 0 && gy < 64 && gx >= 0 && gx < 64) v = inc[gy * 64 + gx];
      smem[idx] = v;
    }
    __syncthreads();
    const float* wc = wb + cc * 9;
    int base = ty * 18 + tx;
    acc += wc[0] * smem[base]      + wc[1] * smem[base + 1]  + wc[2] * smem[base + 2]
         + wc[3] * smem[base + 18] + wc[4] * smem[base + 19] + wc[5] * smem[base + 20]
         + wc[6] * smem[base + 36] + wc[7] * smem[base + 37] + wc[8] * smem[base + 38];
    __syncthreads();
  }
  out[((size_t)b * 128 + o) * NPIX + (ty0 + ty) * 64 + tx0 + tx] = acc;
}

// 10) training-mode BN stats over (B,H,W): stats[ch]=mean, stats[128+ch]=rstd
__global__ __launch_bounds__(256)
void bn_stats_kernel(const float* __restrict__ buf, float* __restrict__ stats) {
  int ch = blockIdx.x, t = threadIdx.x;
  __shared__ float red[CCH];
  float s = 0.f, sq = 0.f;
  for (int b = 0; b < BB; ++b) {
    const float* p = buf + ((size_t)b * 128 + ch) * NPIX;
    for (int i = t; i < NPIX; i += 256) { float v = p[i]; s += v; sq += v * v; }
  }
  float S  = block_sum256(s, red);
  float SQ = block_sum256(sq, red);
  if (t == 0) {
    float m = S * (1.0f / (BB * NPIX));
    float var = SQ * (1.0f / (BB * NPIX)) - m * m;
    stats[ch] = m;
    stats[128 + ch] = rsqrtf(var + 1e-5f);
  }
}

// 11) head: out[b,k,n] = b2[k] + sum_o w2[k,o] * relu(bn(buf[b,o,n]))
__global__ __launch_bounds__(256)
void head_kernel(const float* __restrict__ buf, const float* __restrict__ stats,
                 const float* __restrict__ g, const float* __restrict__ beta,
                 const float* __restrict__ w2, const float* __restrict__ b2,
                 float* __restrict__ out) {
  int n = blockIdx.x * 256 + threadIdx.x;
  int b = blockIdx.y;
  float a0 = b2[0], a1 = b2[1];
  for (int o = 0; o < 128; ++o) {
    float v = buf[((size_t)b * 128 + o) * NPIX + n];
    v = (v - stats[o]) * stats[128 + o] * g[o] + beta[o];
    v = fmaxf(v, 0.f);
    a0 += w2[o] * v;
    a1 += w2[128 + o] * v;
  }
  out[((size_t)b * 2 + 0) * NPIX + n] = a0;
  out[((size_t)b * 2 + 1) * NPIX + n] = a1;
}

// ---------------------------------------------------------------------------
extern "C" void kernel_launch(void* const* d_in, const int* in_sizes, int n_in,
                              void* d_out, int out_size, void* d_ws, size_t ws_size,
                              hipStream_t stream) {
  const float* img_feat   = (const float*)d_in[0];
  const float* txt_feat   = (const float*)d_in[1];
  const float* density    = (const float*)d_in[2];
  const float* tp_w       = (const float*)d_in[3];
  const float* tp_b       = (const float*)d_in[4];
  // d_in[5..8] = wq,bq,wk,bk : mathematically dead (softmax over kv-len 1)
  const float* wv         = (const float*)d_in[9];
  const float* bv         = (const float*)d_in[10];
  const float* wo         = (const float*)d_in[11];
  const float* bo         = (const float*)d_in[12];
  const float* ln_g       = (const float*)d_in[13];
  const float* ln_b       = (const float*)d_in[14];
  const float* op_w       = (const float*)d_in[15];
  const float* op_b       = (const float*)d_in[16];
  const float* hm1_w      = (const float*)d_in[17];
  const float* hm1_b      = (const float*)d_in[18];
  const float* hm_bn_g    = (const float*)d_in[19];
  const float* hm_bn_b    = (const float*)d_in[20];
  const float* hm2_w      = (const float*)d_in[21];
  const float* hm2_b      = (const float*)d_in[22];
  const float* hv1_w      = (const float*)d_in[23];
  const float* hv1_b      = (const float*)d_in[24];
  const float* hv_bn_g    = (const float*)d_in[25];
  const float* hv_bn_b    = (const float*)d_in[26];
  const float* hv2_w      = (const float*)d_in[27];
  const float* hv2_b      = (const float*)d_in[28];
  const float* temperature= (const float*)d_in[29];

  // ---- workspace carve (256B aligned) -------------------------------------
  char* p = (char*)d_ws;
  auto carve = [&](size_t bytes) {
    void* r = (void*)p;
    p += (bytes + 255) & ~(size_t)255;
    return r;
  };
  float* txt_n      = (float*)carve(sizeof(float) * BB * CCH);
  float* attn_const = (float*)carve(sizeof(float) * BB * CCH);
  float* cosb       = (float*)carve(sizeof(float) * BB * NPIX);
  float* q_marg     = (float*)carve(sizeof(float) * BB * NPIX);
  float* p_marg     = (float*)carve(sizeof(float) * BB * NPIX);
  float* u_buf      = (float*)carve(sizeof(float) * BB * NPIX);
  float* vv_buf     = (float*)carve(sizeof(float) * BB * NPIX);
  f16*   sn_h       = (f16*)  carve(sizeof(f16) * (size_t)BB * NPIX * CCH);
  f16*   srcT_h     = (f16*)  carve(sizeof(f16) * (size_t)BB * NPIX * CCH);
  f16*   Kmat       = (f16*)  carve(sizeof(f16) * (size_t)BB * NPIX * NPIX); // 67 MB: L2-resident
  float* fusedb     = (float*)carve(sizeof(float) * (size_t)BB * NPIX * CCH);
  float* hm1buf     = (float*)carve(sizeof(float) * (size_t)BB * 128 * NPIX);
  float* hv1buf     = (float*)carve(sizeof(float) * (size_t)BB * 128 * NPIX);
  float* hm_stats   = (float*)carve(sizeof(float) * 256);
  float* hv_stats   = (float*)carve(sizeof(float) * 256);

  float* out_feat = (float*)d_out;                    // [B,256,64,64]
  float* hm_out   = out_feat + (size_t)BB * CCH * NPIX;
  float* hv_out   = hm_out + (size_t)BB * 2 * NPIX;

  // 1) text projection + constant attention output
  text_kernel<<<BB, 256, 0, stream>>>(txt_feat, tp_w, tp_b, wv, bv, wo, bo,
                                      txt_n, attn_const);
  // 2) residual + LayerNorm -> source / sn / cos
  source_kernel<<<BB * NPIX, 256, 0, stream>>>(img_feat, attn_const, ln_g, ln_b,
                                               txt_n, sn_h, srcT_h, cosb);
  // 3) marginals
  marginals_kernel<<<BB, 256, 0, stream>>>(density, cosb, temperature, q_marg, p_marg);
  // 4) K = exp((sn sn^T - 1)/eps), f16, WMMA  (B*256*16 waves / 8 per block)
  sim_exp_kernel<<<(BB * 256 * 16) / 8, 256, 0, stream>>>(sn_h, Kmat);
  // 5) Sinkhorn (K symmetric -> row reductions both directions)
  fill_kernel<<<(BB * NPIX) / 256, 256, 0, stream>>>(u_buf, 1.0f, BB * NPIX);
  dim3 sgrid(NPIX / 8, BB);
  for (int it = 0; it < 3; ++it) {
    sinkhorn_step<<<sgrid, 256, 0, stream>>>(Kmat, u_buf, q_marg, vv_buf);
    sinkhorn_step<<<sgrid, 256, 0, stream>>>(Kmat, vv_buf, p_marg, u_buf);
  }
  // 6) Ks[m,n] = K[m,n]*u[n]  (in place)
  scale_k_kernel<<<8192, 256, 0, stream>>>(Kmat, u_buf, (size_t)BB << 24);
  // 7) fused[m,c] = vv[m] * (Ks @ source)[m,c]  (WMMA)
  fused_gemm_kernel<<<(BB * 256 * 16) / 8, 256, 0, stream>>>(Kmat, srcT_h, vv_buf, fusedb);
  // 8) out_feat = img + conv1x1(fused^T)
  proj_add_kernel<<<BB * NPIX, 256, 0, stream>>>(img_feat, fusedb, op_w, op_b, out_feat);
  // 9-11) two heads: conv3x3 -> BN(train) -> relu -> conv1x1
  dim3 cgrid(16, 128, BB);
  conv3_kernel<<<cgrid, 256, 0, stream>>>(out_feat, hm1_w, hm1_b, hm1buf);
  conv3_kernel<<<cgrid, 256, 0, stream>>>(out_feat, hv1_w, hv1_b, hv1buf);
  bn_stats_kernel<<<128, 256, 0, stream>>>(hm1buf, hm_stats);
  bn_stats_kernel<<<128, 256, 0, stream>>>(hv1buf, hv_stats);
  dim3 hgrid(NPIX / 256, BB);
  head_kernel<<<hgrid, 256, 0, stream>>>(hm1buf, hm_stats, hm_bn_g, hm_bn_b,
                                         hm2_w, hm2_b, hm_out);
  head_kernel<<<hgrid, 256, 0, stream>>>(hv1buf, hv_stats, hv_bn_g, hv_bn_b,
                                         hv2_w, hv2_b, hv_out);
}
